// GraphNC_81604378624597
// MI455X (gfx1250) — compile-verified, matrix operand order checked by reading, not run on hardware
//
#include <hip/hip_runtime.h>

// GNN edge MLP: out[e] = sigmoid(relu(concat(zi[src[e]], zj[dst[e]]) @ W1 + b1) @ W3 + b3)
// E=600000, D=128, HID=128.  f16 WMMA (f32 accumulate) on gfx1250.

typedef __attribute__((ext_vector_type(16))) _Float16 v16h;
typedef __attribute__((ext_vector_type(8)))  float    v8f;

#define D_FEAT 128
#define HID    128
#define TILE_E 16
#define NFRAG  64   // 8 K-blocks x 8 N-tiles of W1 in B-fragment layout

__global__ __launch_bounds__(256) void gnn_edge_mlp(
    const float* __restrict__ zi, const float* __restrict__ zj,
    const int*   __restrict__ src, const int* __restrict__ dst,
    const float* __restrict__ W1, const float* __restrict__ b1,
    const float* __restrict__ W3, const float* __restrict__ b3,
    float* __restrict__ out, int nEdges, int nTiles)
{
    // 64 KB: per (fragment, lane) 16 contiguous f16 -> 2x ds_load_b128 per B operand
    __shared__ _Float16 sW1[NFRAG * 32 * 16];

    // ---- Cooperative fill: W1 (256x128 f32 row-major) -> f16 B-fragments ----
    // Fragment f = kb*8 + nt.  Lane ln: col = ln&15 (N within tile), K set =
    // {kb*32 + kbase + j} U {kb*32 + kbase + 16 + j}, kbase = (ln>>4)*8, j=0..7.
    for (int idx = threadIdx.x; idx < NFRAG * 32; idx += blockDim.x) {
        const int f  = idx >> 5, ln = idx & 31;
        const int kb = f >> 3,   nt = f & 7;
        const int c  = ln & 15;
        const int kbase = (ln >> 4) << 3;
        _Float16* dp = &sW1[idx << 4];
        #pragma unroll
        for (int j = 0; j < 8; ++j) {
            const int K0 = kb * 32 + kbase + j;
            dp[j]     = (_Float16)W1[(size_t)K0        * HID + nt * 16 + c];
            dp[8 + j] = (_Float16)W1[(size_t)(K0 + 16) * HID + nt * 16 + c];
        }
    }
    __syncthreads();

    const int lane  = threadIdx.x & 31;
    const int col   = lane & 15;     // A: edge-row within tile / BCD: N column
    const int half  = lane >> 4;
    const int kbase = half << 3;

    float b1v[8], w3v[8];
    #pragma unroll
    for (int nt = 0; nt < 8; ++nt) {
        b1v[nt] = b1[nt * 16 + col];
        w3v[nt] = W3[nt * 16 + col];
    }
    const float b3s = b3[0];

    const v16h* sB = (const v16h*)sW1;

    const int waveId = blockIdx.x * (blockDim.x >> 5) + (threadIdx.x >> 5);
    const int nWaves = gridDim.x * (blockDim.x >> 5);

    for (int t = waveId; t < nTiles; t += nWaves) {
        const int e0 = t * TILE_E;
        int e = e0 + col;
        if (e >= nEdges) e = nEdges - 1;           // branchless clamp (keeps EXEC full)
        const int ns = src[e], nd = dst[e];
        const float* pi = zi + (size_t)ns * D_FEAT;
        const float* pj = zj + (size_t)nd * D_FEAT;

        // ---- Gather A fragments: 16 edges x 256 feats, f32 -> f16 ----
        v16h aF[8];
        #pragma unroll
        for (int kb = 0; kb < 8; ++kb) {
            const float* base = (kb < 4) ? (pi + kb * 32 + kbase)
                                         : (pj + (kb - 4) * 32 + kbase);
            const float4 c0 = ((const float4*)base)[0];
            const float4 c1 = ((const float4*)base)[1];
            const float4 c2 = ((const float4*)(base + 16))[0];
            const float4 c3 = ((const float4*)(base + 16))[1];
            v16h a;
            a[0]=(_Float16)c0.x;  a[1]=(_Float16)c0.y;  a[2]=(_Float16)c0.z;  a[3]=(_Float16)c0.w;
            a[4]=(_Float16)c1.x;  a[5]=(_Float16)c1.y;  a[6]=(_Float16)c1.z;  a[7]=(_Float16)c1.w;
            a[8]=(_Float16)c2.x;  a[9]=(_Float16)c2.y;  a[10]=(_Float16)c2.z; a[11]=(_Float16)c2.w;
            a[12]=(_Float16)c3.x; a[13]=(_Float16)c3.y; a[14]=(_Float16)c3.z; a[15]=(_Float16)c3.w;
            aF[kb] = a;
        }

        float p[8];
        #pragma unroll
        for (int r = 0; r < 8; ++r) p[r] = 0.f;

        // ---- Layer 1: 8 N-tiles x 8 K-blocks of v_wmma_f32_16x16x32_f16 ----
        #pragma unroll
        for (int nt = 0; nt < 8; ++nt) {
            // Launder the lane index so the 64 B-fragment LDS loads are NOT
            // loop-invariant: stops MachineLICM hoisting them out of the tile
            // loop (which previously spilled 2KB/lane of B-fragments to
            // scratch and fed every WMMA from scratch_load_b128).
            int lnx = lane;
            asm volatile("" : "+v"(lnx));

            v8f acc;
            #pragma unroll
            for (int r = 0; r < 8; ++r) acc[r] = b1v[nt];   // bias pre-load into C
            #pragma unroll
            for (int kb = 0; kb < 8; ++kb) {
                acc = __builtin_amdgcn_wmma_f32_16x16x32_f16(
                    false, aF[kb],
                    false, sB[(kb * 8 + nt) * 32 + lnx],
                    (short)0, acc, false, false);
            }
            // ---- Fused ReLU + layer-2 partial dot (per-lane column of W3) ----
            const float w3 = w3v[nt];
            #pragma unroll
            for (int r = 0; r < 8; ++r)
                p[r] += __builtin_fmaxf(acc[r], 0.f) * w3;
        }

        // ---- Reduce over the 16 N-lanes of each half-wave ----
        // C/D layout: VGPR r of lane L holds row (r + 8*(L>>4)), col (L&15).
        #pragma unroll
        for (int r = 0; r < 8; ++r) {
            #pragma unroll
            for (int off = 1; off < 16; off <<= 1)
                p[r] += __shfl_xor(p[r], off, 16);
        }

        // ---- Fused bias + sigmoid, store 8 edges per half-wave ----
        if (col == 0) {
            const int eBase = e0 + half * 8;
            if (eBase + 7 < nEdges) {
                float4 o0, o1;
                o0.x = 1.f / (1.f + __expf(-(p[0] + b3s)));
                o0.y = 1.f / (1.f + __expf(-(p[1] + b3s)));
                o0.z = 1.f / (1.f + __expf(-(p[2] + b3s)));
                o0.w = 1.f / (1.f + __expf(-(p[3] + b3s)));
                o1.x = 1.f / (1.f + __expf(-(p[4] + b3s)));
                o1.y = 1.f / (1.f + __expf(-(p[5] + b3s)));
                o1.z = 1.f / (1.f + __expf(-(p[6] + b3s)));
                o1.w = 1.f / (1.f + __expf(-(p[7] + b3s)));
                float4* op = (float4*)(out + eBase);
                op[0] = o0; op[1] = o1;
            } else {
                #pragma unroll
                for (int r = 0; r < 8; ++r) {
                    const int eo = eBase + r;
                    if (eo < nEdges)
                        out[eo] = 1.f / (1.f + __expf(-(p[r] + b3s)));
                }
            }
        }
    }
}

extern "C" void kernel_launch(void* const* d_in, const int* in_sizes, int n_in,
                              void* d_out, int out_size, void* d_ws, size_t ws_size,
                              hipStream_t stream) {
    const float* zi = (const float*)d_in[0];
    const float* zj = (const float*)d_in[1];
    const int*   src = (const int*)d_in[2];
    const int*   dst = (const int*)d_in[3];
    const float* W1 = (const float*)d_in[4];
    const float* b1 = (const float*)d_in[5];
    const float* W3 = (const float*)d_in[6];
    const float* b3 = (const float*)d_in[7];
    float* out = (float*)d_out;

    const int nEdges = in_sizes[2];
    const int nTiles = (nEdges + TILE_E - 1) / TILE_E;
    int blocks = (nTiles + 7) / 8;        // 8 waves per block
    if (blocks > 1024) blocks = 1024;     // persistent grid-stride: amortize W1->LDS fill
    if (blocks < 1) blocks = 1;

    gnn_edge_mlp<<<blocks, 256, 0, stream>>>(zi, zj, src, dst, W1, b1, W3, b3,
                                             out, nEdges, nTiles);
}
